// EdgeUpdate_60120952209607
// MI455X (gfx1250) — matrix operand order repaired and checked by reading, not compiled, not run on hardware
//
#include <hip/hip_runtime.h>
#include <cstdint>

typedef _Float16 half16  __attribute__((ext_vector_type(16)));
typedef _Float16 half8   __attribute__((ext_vector_type(8)));
typedef float    float8  __attribute__((ext_vector_type(8)));
typedef float    float4v __attribute__((ext_vector_type(4)));

#define N_EDGES 500000
#define D_NODE  128
#define D_EDGE  128
#define HIDDEN  64
#define D_IN    384   // 2*D_NODE + D_EDGE
#define M_TILES 2     // 16-edge M-tiles per wave (32 edges/wave)

__device__ __forceinline__ half16 concat8(half8 lo, half8 hi) {
  return __builtin_shufflevector(lo, hi, 0,1,2,3,4,5,6,7,8,9,10,11,12,13,14,15);
}

// A-fragment (16xK rows, row-major f16 in LDS). ISA 16-bit A 16x32 layout:
// lane L: row = L&15, halves 0..7 = K[kb..kb+7], halves 8..15 = K[kb+16..kb+23],
// kb = k0 + (L>=16 ? 8 : 0).
__device__ __forceinline__ half16 ldsA(const _Float16* buf, int ldk, int k0, int lane) {
  const int r  = lane & 15;
  const int kb = k0 + ((lane >> 4) << 3);
  const _Float16* row = buf + r * ldk;
  half8 lo = *(const half8*)(row + kb);
  half8 hi = *(const half8*)(row + kb + 16);
  return concat8(lo, hi);
}

// B-fragment (Kx16), stored as B^T row-major [ncols][ldk] f16 in LDS.
// Dense f16 B layout (matches A): lane L col n=L&15, halves 0..7 = K[kb..kb+7],
// halves 8..15 = K[kb+16..kb+23], kb = k0 + (L>=16 ? 8 : 0).
__device__ __forceinline__ half16 ldsB(const _Float16* buf, int ldk, int ncolbase,
                                       int k0, int lane) {
  const int n  = lane & 15;
  const int ks = k0 + ((lane >> 4) << 3);
  const _Float16* row = buf + (ncolbase + n) * ldk;
  half8 lo = *(const half8*)(row + ks);
  half8 hi = *(const half8*)(row + ks + 16);
  return concat8(lo, hi);
}

// A-fragment directly from a global f32 row (converted to f16).
__device__ __forceinline__ half16 gblA(const float* __restrict__ row, int kb) {
  float4v f0 = *(const float4v*)(row + kb);
  float4v f1 = *(const float4v*)(row + kb + 4);
  float4v f2 = *(const float4v*)(row + kb + 16);
  float4v f3 = *(const float4v*)(row + kb + 20);
  half16 a;
#pragma unroll
  for (int i = 0; i < 4; ++i) {
    a[i]      = (_Float16)f0[i];
    a[4 + i]  = (_Float16)f1[i];
    a[8 + i]  = (_Float16)f2[i];
    a[12 + i] = (_Float16)f3[i];
  }
  return a;
}

__global__ __launch_bounds__(256)
void edge_mlp_wmma(const float* __restrict__ x,
                   const long long* __restrict__ ei,   // [2][E] int64
                   const float* __restrict__ e,
                   const float* __restrict__ W1,       // [384][64]
                   const float* __restrict__ b1,       // [64]
                   const float* __restrict__ W2,       // [64][128]
                   const float* __restrict__ b2,       // [128]
                   float* __restrict__ out) {          // [E][128]
  __shared__ alignas(16) _Float16 sW1T[HIDDEN * D_IN];        // W1^T [64][384], 48 KB
  __shared__ alignas(16) _Float16 sW2T[D_EDGE * HIDDEN];      // W2^T [128][64], 16 KB
  __shared__ alignas(16) _Float16 sH[8][M_TILES * 16 * HIDDEN]; // h staging, 32 KB

  const int tid = threadIdx.x;

  // One-time per block: convert weights to f16, transposed, into LDS.
  for (int i = tid; i < HIDDEN * D_IN; i += 256) {
    const int k = i >> 6, n = i & 63;          // W1 row-major [k][n]
    sW1T[n * D_IN + k] = (_Float16)W1[i];
  }
  for (int i = tid; i < D_EDGE * HIDDEN; i += 256) {
    const int k = i >> 7, n = i & 127;         // W2 row-major [k][n]
    sW2T[n * HIDDEN + k] = (_Float16)W2[i];
  }
  __syncthreads();

  const int wave = tid >> 5;                   // wave32
  const int lane = tid & 31;
  const long long tileBase = ((long long)blockIdx.x * 8 + wave) * (M_TILES * 16);
  if (tileBase >= N_EDGES) return;             // E % 32 == 0 -> surviving jobs full

  // Per-lane edge row pointers for each of the two M-tiles.
  const int r = lane & 15;
  const float* xsrc[M_TILES];
  const float* xdst[M_TILES];
  const float* erow[M_TILES];
#pragma unroll
  for (int t = 0; t < M_TILES; ++t) {
    long long edge = tileBase + t * 16 + r;
    if (edge >= N_EDGES) edge = N_EDGES - 1;   // defensive clamp; keeps EXEC full
    const int si = (int)ei[edge];
    const int di = (int)ei[N_EDGES + edge];
    xsrc[t] = x + (size_t)si * D_NODE;
    xdst[t] = x + (size_t)di * D_NODE;
    erow[t] = e + (size_t)edge * D_EDGE;
  }

  // ---- GEMM1: [32x384] x [384x64] -> [32x64], f32 accum.
  // Each B-fragment from LDS feeds both M-tiles (halves DS traffic/FLOP).
  float8 acc[M_TILES][4] = {};
#pragma unroll
  for (int kt = 0; kt < 12; ++kt) {
    const int k0  = kt * 32;
    const int seg = k0 >> 7;                   // 0: x[src], 1: x[dst], 2: e
    const int off = k0 & 127;
    const int kb  = off + ((lane >> 4) << 3);
    half16 a[M_TILES];
#pragma unroll
    for (int t = 0; t < M_TILES; ++t) {
      const float* rowp = (seg == 0) ? xsrc[t] : (seg == 1) ? xdst[t] : erow[t];
      a[t] = gblA(rowp, kb);
    }
#pragma unroll
    for (int nt = 0; nt < 4; ++nt) {
      half16 b = ldsB(sW1T, D_IN, nt * 16, k0, lane);
#pragma unroll
      for (int t = 0; t < M_TILES; ++t) {
        acc[t][nt] = __builtin_amdgcn_wmma_f32_16x16x32_f16(
            false, a[t], false, b, (short)0, acc[t][nt], false, false);
      }
    }
  }

  // Bias + ReLU; stage h (f16) in per-wave LDS to transpose C-layout -> A-layout.
  _Float16* hbuf = sH[wave];
  const int nh    = lane & 15;
  const int mbase = (lane >> 4) << 3;          // C layout: VGPR i -> M = mbase + i
#pragma unroll
  for (int nt = 0; nt < 4; ++nt) {
    const int col = nt * 16 + nh;
    const float bv = b1[col];
#pragma unroll
    for (int t = 0; t < M_TILES; ++t) {
#pragma unroll
      for (int i = 0; i < 8; ++i) {
        float v = acc[t][nt][i] + bv;
        v = v > 0.0f ? v : 0.0f;
        hbuf[(t * 16 + mbase + i) * HIDDEN + col] = (_Float16)v;
      }
    }
  }

  // ---- GEMM2: [32x64] x [64x128] -> [32x128]; shared B-fragments again.
  float8 acc2[M_TILES][8] = {};
#pragma unroll
  for (int kt = 0; kt < 2; ++kt) {
    const int k0 = kt * 32;
    half16 a[M_TILES];
#pragma unroll
    for (int t = 0; t < M_TILES; ++t)
      a[t] = ldsA(hbuf + t * 16 * HIDDEN, HIDDEN, k0, lane);
#pragma unroll
    for (int nt = 0; nt < 8; ++nt) {
      half16 b = ldsB(sW2T, HIDDEN, nt * 16, k0, lane);
#pragma unroll
      for (int t = 0; t < M_TILES; ++t) {
        acc2[t][nt] = __builtin_amdgcn_wmma_f32_16x16x32_f16(
            false, a[t], false, b, (short)0, acc2[t][nt], false, false);
      }
    }
  }

  // Epilogue: out = e + (h@W2 + b2), f32 residual read from global (exact).
#pragma unroll
  for (int t = 0; t < M_TILES; ++t) {
    const long long base = tileBase + t * 16;
#pragma unroll
    for (int nt = 0; nt < 8; ++nt) {
      const int col = nt * 16 + nh;
      const float bv = b2[col];
#pragma unroll
      for (int i = 0; i < 8; ++i) {
        const long long m = base + mbase + i;
        if (m < N_EDGES) {
          const size_t idx = (size_t)m * D_EDGE + col;
          out[idx] = e[idx] + acc2[t][nt][i] + bv;
        }
      }
    }
  }
}

extern "C" void kernel_launch(void* const* d_in, const int* in_sizes, int n_in,
                              void* d_out, int out_size, void* d_ws, size_t ws_size,
                              hipStream_t stream) {
  const float*     x  = (const float*)d_in[0];
  const long long* ei = (const long long*)d_in[1];
  const float*     e  = (const float*)d_in[2];
  const float*     W1 = (const float*)d_in[3];
  const float*     b1 = (const float*)d_in[4];
  const float*     W2 = (const float*)d_in[5];
  const float*     b2 = (const float*)d_in[6];
  float* out = (float*)d_out;

  const int jobs   = (N_EDGES + M_TILES * 16 - 1) / (M_TILES * 16); // 15625 wave-jobs
  const int blocks = (jobs + 7) / 8;                                // 8 waves/block
  edge_mlp_wmma<<<blocks, 256, 0, stream>>>(x, ei, e, W1, b1, W2, b2, out);

  (void)in_sizes; (void)n_in; (void)out_size; (void)d_ws; (void)ws_size;
}
